// MLACpu_52828097740986
// MI455X (gfx1250) — matrix operand order repaired and checked by reading, not compiled, not run on hardware
//
#include <hip/hip_runtime.h>
#include <hip/hip_bf16.h>
#include <cstddef>
#include <cstdint>

typedef __bf16 bf16;
typedef __attribute__((ext_vector_type(16))) bf16  v16bf;
typedef __attribute__((ext_vector_type(8)))  float v8f;
typedef unsigned int u32x4 __attribute__((ext_vector_type(4)));
typedef int          s32x8 __attribute__((ext_vector_type(8)));
typedef int          s32x4 __attribute__((ext_vector_type(4)));

#define B_    4
#define S_    1024
#define H_    32
#define NOPE_ 128
#define ROPE_ 64
#define VDIM_ 128
#define DQK_  192        // NOPE + ROPE
#define QL_   1536
#define KVL_  512
#define SCALE_ 0.07216878364870322f  // 1/sqrt(192)
#define NEG_INF_ (-3.0e38f)

__device__ __forceinline__ bf16 f2bf(float f) {
  unsigned u = __builtin_bit_cast(unsigned, f);
  unsigned r = (u + 0x7FFFu + ((u >> 16) & 1u)) >> 16;   // round-to-nearest-even
  unsigned short h = (unsigned short)r;
  return __builtin_bit_cast(bf16, h);
}

__device__ __forceinline__ v8f wmma_bf16(v16bf a, v16bf b, v8f c) {
  return __builtin_amdgcn_wmma_f32_16x16x32_bf16(false, a, false, b, (short)0, c,
                                                 false, false);
}

// ---- fragment builders (layouts per CDNA5 ISA §7.12.2, wave32) ----

// A fragment 16x32 bf16 from LDS tile As[128][32]; wave owns rows [wm, wm+16).
__device__ __forceinline__ v16bf lds_a_frag(const bf16* As, int wm, int lane) {
  int m  = wm + (lane & 15);
  int kg = (lane >> 4) * 8;
  const bf16* p0 = As + m * 32 + kg;
  const bf16* p1 = p0 + 16;
  v16bf a;
#pragma unroll
  for (int i = 0; i < 8; ++i) { a[i] = p0[i]; a[8 + i] = p1[i]; }
  return a;
}

// B fragment 32(K)x16(N): lane n = lane&15, K = (lane>>4)*8 + e, and 16 + (lane>>4)*8 + e.
__device__ __forceinline__ v16bf lds_b_frag(const bf16* Bs, int t, int lane) {
  const bf16* p = Bs + (t * 16 + (lane & 15)) * 32 + (lane >> 4) * 8;
  v16bf b;
#pragma unroll
  for (int i = 0; i < 8; ++i) { b[i] = p[i]; b[8 + i] = p[16 + i]; }
  return b;
}

// B fragment from an LDS tile with row stride `stride` elements: row nbase+lane&15,
// 8 elements at column kc + (lane>>4)*8 and 8 at kc + 16 + (lane>>4)*8.
__device__ __forceinline__ v16bf lds_b_frag_s(const bf16* Ts, int nbase, int stride,
                                              int kc, int lane) {
  const bf16* p = Ts + (nbase + (lane & 15)) * stride + kc + (lane >> 4) * 8;
  v16bf b;
#pragma unroll
  for (int i = 0; i < 8; ++i) { b[i] = p[i]; b[8 + i] = p[16 + i]; }
  return b;
}

// ---- Tensor Data Mover: 2D tile load (bf16) global -> LDS (ISA 08_async_tensor §8) ----
// D# group0: count=1 | lds_addr | global_addr[56:0] | type=2
// D# group1: data_size=1(2B) | tensor_dim0/1 | tile_dim0/1 | tensor_dim0_stride
// 6-arg builtin form: (g0, g1, g2, g3, g_extra, cpol); groups 2/3 zero (2-D tile).
__device__ __forceinline__ void tdm_load_2d_bf16(const bf16* lds_dst,
                                                 const bf16* gsrc,
                                                 uint32_t row_len, uint32_t rows,
                                                 uint32_t row_stride,
                                                 uint32_t tile_w, uint32_t tile_h) {
  uint64_t ga = (uint64_t)(uintptr_t)gsrc;
  uint32_t lds_off = (uint32_t)(uintptr_t)(void*)lds_dst;   // low 32 bits: LDS byte offset
  u32x4 g0;
  g0[0] = 1u;                                   // count=1, user descriptor
  g0[1] = lds_off;                              // lds_addr (bytes)
  g0[2] = (uint32_t)ga;                         // global_addr[31:0]
  g0[3] = (uint32_t)(ga >> 32) | (2u << 30);    // global_addr[56:32] | type=2
  s32x8 g1;
  g1[0] = (int)(1u << 16);                                  // data_size = 2 bytes
  g1[1] = (int)((row_len & 0xFFFFu) << 16);                 // tensor_dim0[15:0]
  g1[2] = (int)((row_len >> 16) | ((rows & 0xFFFFu) << 16));// dim0[31:16]|dim1[15:0]
  g1[3] = (int)((rows >> 16) | (tile_w << 16));             // dim1[31:16]|tile_dim0
  g1[4] = (int)(tile_h & 0xFFFFu);                          // tile_dim1 (tile_dim2=0)
  g1[5] = (int)row_stride;                                  // tensor_dim0_stride[31:0]
  g1[6] = 0;
  g1[7] = 0;
  s32x4 z4 = {0, 0, 0, 0};
  s32x8 z8 = {0, 0, 0, 0, 0, 0, 0, 0};
  __builtin_amdgcn_tensor_load_to_lds(g0, g1, z4, z4, z8, 0);
}

// ============================ GEMM 1: q = Q @ WUQ^T + RoPE(q_pe) ============================
__global__ __launch_bounds__(256) void gemm_q_up(const float* __restrict__ Q,
                                                 const float* __restrict__ WUQ,
                                                 const float* __restrict__ cosb,
                                                 const float* __restrict__ sinb,
                                                 bf16* __restrict__ qf) {
  __shared__ bf16 As[128 * 32];
  __shared__ bf16 Bs[128 * 32];
  __shared__ float Ep[8][256];
  const int tid = threadIdx.x, lane = tid & 31, w = tid >> 5;
  const int mb = blockIdx.y * 128, nb = blockIdx.x * 128;

  v8f acc[8];
#pragma unroll
  for (int t = 0; t < 8; ++t)
#pragma unroll
    for (int e = 0; e < 8; ++e) acc[t][e] = 0.f;

  for (int kb = 0; kb < QL_; kb += 32) {
    __syncthreads();
    for (int i = tid; i < 1024; i += 256) {
      int r = i >> 3, c = (i & 7) << 2;
      float4 f = *(const float4*)(Q + (size_t)(mb + r) * QL_ + kb + c);
      bf16* d = As + r * 32 + c;
      d[0] = f2bf(f.x); d[1] = f2bf(f.y); d[2] = f2bf(f.z); d[3] = f2bf(f.w);
      if (kb + 32 < QL_) __builtin_prefetch(Q + (size_t)(mb + r) * QL_ + kb + 32 + c, 0, 1);
    }
    for (int i = tid; i < 1024; i += 256) {
      int r = i >> 3, c = (i & 7) << 2;
      float4 f = *(const float4*)(WUQ + (size_t)(nb + r) * QL_ + kb + c);
      bf16* d = Bs + r * 32 + c;
      d[0] = f2bf(f.x); d[1] = f2bf(f.y); d[2] = f2bf(f.z); d[3] = f2bf(f.w);
      if (kb + 32 < QL_) __builtin_prefetch(WUQ + (size_t)(nb + r) * QL_ + kb + 32 + c, 0, 1);
    }
    __syncthreads();
    v16bf a = lds_a_frag(As, w * 16, lane);
#pragma unroll
    for (int t = 0; t < 8; ++t) {
      v16bf b = lds_b_frag(Bs, t, lane);
      acc[t] = wmma_bf16(a, b, acc[t]);
    }
  }

  const int g = lane >> 4, nl = lane & 15;
#pragma unroll
  for (int t = 0; t < 8; ++t) {
    int ntile = nb + t * 16;          // 16-wide tile never straddles head/region bounds
    int h     = ntile / DQK_;
    int dbase = ntile % DQK_;
    if (dbase < NOPE_) {              // q_nope: direct store
      int d = dbase + nl;
#pragma unroll
      for (int r = 0; r < 8; ++r) {
        int row = mb + w * 16 + r + 8 * g;
        int b = row >> 10, s = row & (S_ - 1);
        qf[(((size_t)(b * H_ + h)) * S_ + s) * DQK_ + d] = f2bf(acc[t][r] * SCALE_);
      }
    } else {                          // q_pe: pair-mix via per-wave LDS tile
      float* ep = Ep[w];
#pragma unroll
      for (int r = 0; r < 8; ++r) ep[(r + 8 * g) * 16 + nl] = acc[t][r];
      int jb = (dbase - NOPE_) >> 1;  // tile covers rope pairs j = jb..jb+7
      for (int idx = lane; idx < 128; idx += 32) {
        int rr = idx >> 3, p = idx & 7;
        float x0 = ep[rr * 16 + 2 * p];
        float x1 = ep[rr * 16 + 2 * p + 1];
        int row = mb + w * 16 + rr;
        int b = row >> 10, s = row & (S_ - 1);
        int j = jb + p;
        float c0 = cosb[(size_t)row * ROPE_ + j];
        float s0 = sinb[(size_t)row * ROPE_ + j];
        float c1 = cosb[(size_t)row * ROPE_ + j + 32];
        float s1 = sinb[(size_t)row * ROPE_ + j + 32];
        size_t base = (((size_t)(b * H_ + h)) * S_ + s) * DQK_;
        qf[base + NOPE_ + j]      = f2bf((x0 * c0 - x1 * s0) * SCALE_);
        qf[base + NOPE_ + 32 + j] = f2bf((x1 * c1 + x0 * s1) * SCALE_);
      }
    }
  }
}

// ============================ GEMM 2: kvu = KV @ WUKV^T ============================
__global__ __launch_bounds__(256) void gemm_kv_up(const float* __restrict__ KV,
                                                  const float* __restrict__ WUKV,
                                                  bf16* __restrict__ kf,
                                                  bf16* __restrict__ vtp) {
  __shared__ bf16 As[128 * 32];
  __shared__ bf16 Bs[128 * 32];
  const int tid = threadIdx.x, lane = tid & 31, w = tid >> 5;
  const int mb = blockIdx.y * 128, nb = blockIdx.x * 128;

  v8f acc[8];
#pragma unroll
  for (int t = 0; t < 8; ++t)
#pragma unroll
    for (int e = 0; e < 8; ++e) acc[t][e] = 0.f;

  for (int kb = 0; kb < KVL_; kb += 32) {
    __syncthreads();
    for (int i = tid; i < 1024; i += 256) {
      int r = i >> 3, c = (i & 7) << 2;
      float4 f = *(const float4*)(KV + (size_t)(mb + r) * KVL_ + kb + c);
      bf16* d = As + r * 32 + c;
      d[0] = f2bf(f.x); d[1] = f2bf(f.y); d[2] = f2bf(f.z); d[3] = f2bf(f.w);
    }
    for (int i = tid; i < 1024; i += 256) {
      int r = i >> 3, c = (i & 7) << 2;
      float4 f = *(const float4*)(WUKV + (size_t)(nb + r) * KVL_ + kb + c);
      bf16* d = Bs + r * 32 + c;
      d[0] = f2bf(f.x); d[1] = f2bf(f.y); d[2] = f2bf(f.z); d[3] = f2bf(f.w);
    }
    __syncthreads();
    v16bf a = lds_a_frag(As, w * 16, lane);
#pragma unroll
    for (int t = 0; t < 8; ++t) {
      v16bf b = lds_b_frag(Bs, t, lane);
      acc[t] = wmma_bf16(a, b, acc[t]);
    }
  }

  const int g = lane >> 4, nl = lane & 15;
#pragma unroll
  for (int t = 0; t < 8; ++t) {
    int n = nb + t * 16 + nl;
    int h = n >> 8, d = n & 255;
#pragma unroll
    for (int r = 0; r < 8; ++r) {
      int row = mb + w * 16 + r + 8 * g;
      int b = row >> 10, s = row & (S_ - 1);
      if (d < NOPE_) {
        kf[(((size_t)(b * H_ + h)) * S_ + s) * DQK_ + d] = f2bf(acc[t][r]);
      } else {
        vtp[(((size_t)(b * H_ + h)) * VDIM_ + (d - NOPE_)) * S_ + s] = f2bf(acc[t][r]);
      }
    }
  }
}

// ============================ RoPE(PE) broadcast into k_full ============================
__global__ __launch_bounds__(128) void rope_k(const float* __restrict__ PE,
                                              const float* __restrict__ cosb,
                                              const float* __restrict__ sinb,
                                              bf16* __restrict__ kf) {
  const int t = blockIdx.x;                 // 0..B*S-1
  __shared__ float vals[ROPE_];
  const int tid = threadIdx.x;
  if (tid < ROPE_) {
    const float* x = PE + (size_t)t * ROPE_;
    float c = cosb[(size_t)t * ROPE_ + tid];
    float s = sinb[(size_t)t * ROPE_ + tid];
    float v;
    if (tid < 32) v = x[2 * tid] * c - x[2 * tid + 1] * s;
    else { int j = tid - 32; v = x[2 * j + 1] * c + x[2 * j] * s; }
    vals[tid] = v;
  }
  __syncthreads();
  const int b = t >> 10, s = t & (S_ - 1);
  for (int idx = tid; idx < H_ * ROPE_; idx += 128) {
    int h = idx >> 6, j = idx & 63;
    kf[(((size_t)(b * H_ + h)) * S_ + s) * DQK_ + NOPE_ + j] = f2bf(vals[j]);
  }
}

// ============================ Flash attention (causal, TDM-staged K/V) ============================
// grid.x = S/128 query blocks, grid.y = B*H. 8 waves/block, wave = 16 query rows.
// K tile (32x192, 12KB) and V^T tile (128x32, 8KB) are double-buffered in LDS via
// the Tensor Data Mover; wave 0 issues descriptors and waits on TENSORcnt.
__global__ __launch_bounds__(256) void mla_attn(const bf16* __restrict__ qf,
                                                const bf16* __restrict__ kf,
                                                const bf16* __restrict__ vtp,
                                                float* __restrict__ out) {
  __shared__ bf16 Kt[2][32 * DQK_];    // 2 x 12KB
  __shared__ bf16 Vt[2][VDIM_ * 32];   // 2 x 8KB
  __shared__ float Pt[8][16 * 32];     // per-wave P staging

  const int tid = threadIdx.x, lane = tid & 31, w = tid >> 5;
  const int g = lane >> 4, nl = lane & 15;
  const int bh = blockIdx.y, b = bh >> 5, h = bh & 31;
  const int qbase = blockIdx.x * 128 + w * 16;

  const bf16* qhead = qf + ((size_t)bh * S_) * DQK_;
  const bf16* khead = kf + ((size_t)bh * S_) * DQK_;
  const bf16* vhead = vtp + (size_t)bh * VDIM_ * S_;

  // Q fragments: 6 chunks of K=32 along d=192, rows [qbase, qbase+16)
  v16bf qa[6];
  {
    const bf16* qp = qhead + (size_t)(qbase + nl) * DQK_;
#pragma unroll
    for (int c = 0; c < 6; ++c) {
      int k0 = c * 32 + 8 * g;
#pragma unroll
      for (int i = 0; i < 8; ++i) { qa[c][i] = qp[k0 + i]; qa[c][8 + i] = qp[k0 + 16 + i]; }
    }
  }

  v8f O[8];
#pragma unroll
  for (int u = 0; u < 8; ++u)
#pragma unroll
    for (int e = 0; e < 8; ++e) O[u][e] = 0.f;
  float mi[8], li[8];
#pragma unroll
  for (int r = 0; r < 8; ++r) { mi[r] = NEG_INF_; li[r] = 0.f; }

  float* P = Pt[w];
  const int KT = blockIdx.x * 4 + 3;       // block-wide causal tile bound

  // prologue: stream tile 0 into buffer 0
  if (w == 0) {
    tdm_load_2d_bf16(Kt[0], khead, DQK_, 32, DQK_, DQK_, 32);
    tdm_load_2d_bf16(Vt[0], vhead, S_, VDIM_, S_, 32, VDIM_);
  }

  for (int kt = 0; kt <= KT; ++kt) {
    const int j0 = kt * 32;
    const int buf = kt & 1;
    if (w == 0) {
      if (kt < KT) {   // stream tile kt+1 into the other buffer, then wait for tile kt
        const int j1 = j0 + 32;
        tdm_load_2d_bf16(Kt[buf ^ 1], khead + (size_t)j1 * DQK_, DQK_, 32, DQK_, DQK_, 32);
        tdm_load_2d_bf16(Vt[buf ^ 1], vhead + j1, S_, VDIM_, S_, 32, VDIM_);
        __builtin_amdgcn_s_wait_tensorcnt(2);   // in-order: tile kt pair complete
      } else {
        __builtin_amdgcn_s_wait_tensorcnt(0);
      }
    }
    __syncthreads();   // tile kt visible to all waves

    if (j0 <= qbase + 15) {   // this wave still has unmasked keys in tile kt
      // scores: two 16x16 subtiles, K-dim 192 -> 6 WMMAs each (B frags from LDS)
      v8f S0, S1;
#pragma unroll
      for (int e = 0; e < 8; ++e) { S0[e] = 0.f; S1[e] = 0.f; }
#pragma unroll
      for (int c = 0; c < 6; ++c)
        S0 = wmma_bf16(qa[c], lds_b_frag_s(Kt[buf], 0, DQK_, c * 32, lane), S0);
#pragma unroll
      for (int c = 0; c < 6; ++c)
        S1 = wmma_bf16(qa[c], lds_b_frag_s(Kt[buf], 16, DQK_, c * 32, lane), S1);

      // online softmax, row m = r + 8*g, key = j0 [+16] + nl
#pragma unroll
      for (int r = 0; r < 8; ++r) {
        int q = qbase + r + 8 * g;
        bool msk0 = (j0 + nl) > q;
        bool msk1 = (j0 + 16 + nl) > q;
        float s0v = msk0 ? NEG_INF_ : S0[r];
        float s1v = msk1 ? NEG_INF_ : S1[r];
        float mx = fmaxf(s0v, s1v);
#pragma unroll
        for (int off = 1; off < 16; off <<= 1) mx = fmaxf(mx, __shfl_xor(mx, off, 32));
        float mn = fmaxf(mi[r], mx);
        float p0 = msk0 ? 0.f : __expf(s0v - mn);
        float p1 = msk1 ? 0.f : __expf(s1v - mn);
        float al = __expf(mi[r] - mn);
        float rs = p0 + p1;
#pragma unroll
        for (int off = 1; off < 16; off <<= 1) rs += __shfl_xor(rs, off, 32);
        li[r] = li[r] * al + rs;
        mi[r] = mn;
#pragma unroll
        for (int u = 0; u < 8; ++u) O[u][r] *= al;
        int m = r + 8 * g;
        P[m * 32 + nl]      = p0;
        P[m * 32 + 16 + nl] = p1;
      }

      // P: C-layout -> A-fragment (same-wave LDS round trip, in-order per wave)
      v16bf pa;
      {
        const float* pp = P + nl * 32;
        int k0 = 8 * g;
#pragma unroll
        for (int i = 0; i < 8; ++i) {
          pa[i]     = f2bf(pp[k0 + i]);
          pa[8 + i] = f2bf(pp[k0 + 16 + i]);
        }
      }

      // PV: O[16x128] += P[16x32] x V[32x128]  (V^T tile in LDS, contiguous frags)
#pragma unroll
      for (int u = 0; u < 8; ++u)
        O[u] = wmma_bf16(pa, lds_b_frag_s(Vt[buf], u * 16, 32, 0, lane), O[u]);
    }
    __syncthreads();   // all waves done with buffer kt before it is overwritten
  }

  // normalize and store fp32 out (b, q, h, d)
#pragma unroll
  for (int u = 0; u < 8; ++u) {
#pragma unroll
    for (int r = 0; r < 8; ++r) {
      int q = qbase + r + 8 * g;
      float o = O[u][r] / li[r];
      out[(((size_t)b * S_ + q) * H_ + h) * VDIM_ + u * 16 + nl] = o;
    }
  }
}

extern "C" void kernel_launch(void* const* d_in, const int* in_sizes, int n_in,
                              void* d_out, int out_size, void* d_ws, size_t ws_size,
                              hipStream_t stream) {
  (void)in_sizes; (void)n_in; (void)out_size; (void)ws_size;
  const float* Q    = (const float*)d_in[0];
  const float* KV   = (const float*)d_in[1];
  const float* PE   = (const float*)d_in[2];
  const float* WUQ  = (const float*)d_in[3];
  const float* WUKV = (const float*)d_in[4];
  const float* cosb = (const float*)d_in[5];
  const float* sinb = (const float*)d_in[6];
  float* out = (float*)d_out;

  const size_t qf_elems = (size_t)B_ * H_ * S_ * DQK_;   // 24M bf16
  const size_t kf_elems = qf_elems;
  bf16* qf = (bf16*)d_ws;
  bf16* kf = qf + qf_elems;
  bf16* vt = kf + kf_elems;                              // B*H*128*S bf16

  gemm_q_up <<<dim3(48, 32), 256, 0, stream>>>(Q, WUQ, cosb, sinb, qf);
  gemm_kv_up<<<dim3(64, 32), 256, 0, stream>>>(KV, WUKV, kf, vt);
  rope_k    <<<dim3(B_ * S_), 128, 0, stream>>>(PE, cosb, sinb, kf);
  mla_attn  <<<dim3(S_ / 128, B_ * H_), 256, 0, stream>>>(qf, kf, vt, out);
}
